// HyperMLPModel_764504179420
// MI455X (gfx1250) — compile-verified, matrix operand order checked by reading, not run on hardware
//
#include <hip/hip_runtime.h>
#include <hip/hip_bf16.h>

// ---------------------------------------------------------------------------
// HyperMLP forward on gfx1250 (MI455X).
// Compute-bound (~0.5 TFLOP): all GEMMs on v_wmma_f32_16x16x32_bf16.
// Activations bf16 in memory (halves A-side L2 traffic); weights f32 converted
// to bf16 while staging into LDS; f32 accumulate. Double-buffered LDS with
// async global->LDS copies for the A tile (ASYNCcnt) when available.
// ---------------------------------------------------------------------------

typedef __attribute__((ext_vector_type(16))) __bf16 v16bf;
typedef __attribute__((ext_vector_type(8)))  float  v8f;

#define BN   128
#define BK   32
#define ASTR 40   // LDS row stride (ushorts) = 80B: 16B-aligned AND conflict-free
#define BSTR 40
#define FLAG_BIAS 1
#define FLAG_GELU 2
#define FLAG_RES  4
#define FLAG_OBF  8   // write bf16 output

#if defined(__AMDGCN__) && __has_builtin(__builtin_amdgcn_global_load_async_to_lds_b128)
#define HAVE_ASYNC_LDS 1
typedef int v4i_ __attribute__((vector_size(16)));                 // GCC-style int4
typedef __attribute__((address_space(1))) v4i_* gv4i_p;            // global src
typedef __attribute__((address_space(3))) v4i_* lv4i_p;            // LDS dst
#else
#define HAVE_ASYNC_LDS 0
#endif

__device__ __forceinline__ void wait_async_then_barrier() {
#if HAVE_ASYNC_LDS
#if __has_builtin(__builtin_amdgcn_s_wait_asynccnt)
  __builtin_amdgcn_s_wait_asynccnt(0);
#else
  asm volatile("s_wait_asynccnt 0" ::: "memory");
#endif
#endif
  __syncthreads();
}

__device__ __forceinline__ unsigned short f2bfu(float f) {
  unsigned int u = __float_as_uint(f);
  u += 0x7FFFu + ((u >> 16) & 1u);      // round-to-nearest-even
  return (unsigned short)(u >> 16);
}
__device__ __forceinline__ __bf16 bfbits(unsigned short s) {
  union { unsigned short u; __bf16 b; } c; c.u = s; return c.b;
}
__device__ __forceinline__ float gelu_tanh(float x) {
  float x3 = x * x * x;
  return 0.5f * x * (1.0f + tanhf(0.7978845608028654f * (x + 0.044715f * x3)));
}

// ---------------------------------------------------------------------------
__global__ void embed_kernel(const int* __restrict__ ids,
                             const float* __restrict__ emb,
                             float* __restrict__ x, int D) {
  int row = blockIdx.x;
  int id = ids[row];
  const float4* src = (const float4*)(emb + (size_t)id * D);
  float4* dst = (float4*)(x + (size_t)row * D);
  for (int i = threadIdx.x; i < D / 4; i += blockDim.x) dst[i] = src[i];
}

// ---------------------------------------------------------------------------
// LayerNorm over D=1024 -> bf16 out. One 256-thread block per row.
// ---------------------------------------------------------------------------
__global__ __launch_bounds__(256) void layernorm_bf16_kernel(
    const float* __restrict__ X, const float* __restrict__ w,
    const float* __restrict__ b, unsigned short* __restrict__ Y) {
  __shared__ float red[256];
  const int tid = threadIdx.x;
  const size_t base = (size_t)blockIdx.x * 1024;
  float4 v = *(const float4*)(X + base + tid * 4);

  red[tid] = v.x + v.y + v.z + v.w;
  __syncthreads();
  for (int o = 128; o > 0; o >>= 1) {
    if (tid < o) red[tid] += red[tid + o];
    __syncthreads();
  }
  float mu = red[0] * (1.0f / 1024.0f);
  __syncthreads();

  float dx = v.x - mu, dy = v.y - mu, dz = v.z - mu, dw = v.w - mu;
  red[tid] = dx * dx + dy * dy + dz * dz + dw * dw;
  __syncthreads();
  for (int o = 128; o > 0; o >>= 1) {
    if (tid < o) red[tid] += red[tid + o];
    __syncthreads();
  }
  float inv = rsqrtf(red[0] * (1.0f / 1024.0f) + 1e-5f);

  float4 wv = *(const float4*)(w + tid * 4);
  float4 bv = *(const float4*)(b + tid * 4);
  unsigned short* yp = Y + base + tid * 4;
  yp[0] = f2bfu(dx * inv * wv.x + bv.x);
  yp[1] = f2bfu(dy * inv * wv.y + bv.y);
  yp[2] = f2bfu(dz * inv * wv.z + bv.z);
  yp[3] = f2bfu(dw * inv * wv.w + bv.w);
}

// ---------------------------------------------------------------------------
// M = (P @ Q^T) * tril -> bf16. Tiny (512x512x64).
// ---------------------------------------------------------------------------
__global__ __launch_bounds__(256) void pqmask_kernel(
    const float* __restrict__ P, const float* __restrict__ Q,
    unsigned short* __restrict__ Mo, int S, int R) {
  int s = blockIdx.x * 16 + (threadIdx.x & 15);
  int t = blockIdx.y * 16 + (threadIdx.x >> 4);
  float acc = 0.0f;
  if (s <= t) {
    const float* p = P + (size_t)t * R;
    const float* q = Q + (size_t)s * R;
    #pragma unroll 8
    for (int k = 0; k < R; ++k) acc += p[k] * q[k];
  }
  Mo[(size_t)t * S + s] = f2bfu(acc);
}

// ---------------------------------------------------------------------------
// GEMM: out = epilogue(A_bf16[M,K] @ B_f32[K,N]).
// Block tile (64*MI) x 128, 256 threads = 8 waves (4M x 2N), wave tile
// (MI*16) x 64 = MI x 4 WMMA tiles. Double-buffered LDS; A staged with
// async global->LDS b128 when available; B staged f32->bf16 transposed.
// ---------------------------------------------------------------------------
template <int MI>
__global__ __launch_bounds__(256) void gemm_bf16_wmma(
    const unsigned short* __restrict__ A, unsigned long long sA,
    const float* __restrict__ Bm, unsigned long long sB,
    const float* __restrict__ bias,
    const float* __restrict__ Res,
    float* __restrict__ Cf, unsigned short* __restrict__ Ch,
    unsigned long long sC,
    int N, int K, int flags) {
  constexpr int BMt = 64 * MI;
  constexpr int AITS = (BMt * 4) / 256;   // A 16B-chunks per thread
  constexpr int BITS = 4;                 // B float4 chunks per thread
  __shared__ unsigned short As[2][BMt * ASTR];
  __shared__ unsigned short Bs[2][BN * BSTR];

  const int tid = threadIdx.x;
  const int lane = tid & 31;
  const int wave = tid >> 5;
  const int wm = wave >> 1;   // 0..3
  const int wn = wave & 1;    // 0..1
  const int row0 = blockIdx.y * BMt;
  const int col0 = blockIdx.x * BN;

  const unsigned short* Ab = A + (size_t)blockIdx.z * sA;
  const float* Bb = Bm + (size_t)blockIdx.z * sB;

  // per-thread staging cursors (stepped by BK each K-step: no 64-bit muls in loop)
  const unsigned short* apg[AITS];
  int aofs[AITS];
  #pragma unroll
  for (int it = 0; it < AITS; ++it) {
    int idx = tid + it * 256;
    int r = idx >> 2, c = (idx & 3) * 8;
    apg[it] = Ab + (size_t)(row0 + r) * K + c;
    aofs[it] = r * ASTR + c;
  }
  const float* bpg[BITS];
  int brow[BITS], bcol[BITS];
  #pragma unroll
  for (int it = 0; it < BITS; ++it) {
    int idx = (tid + it * 256) * 4;
    int r = idx >> 7, c = idx & 127;    // k-row, n-col
    bpg[it] = Bb + (size_t)r * N + (col0 + c);
    brow[it] = r; bcol[it] = c;
  }

  auto stage = [&](int buf) {
    #pragma unroll
    for (int it = 0; it < AITS; ++it) {
#if HAVE_ASYNC_LDS
      __builtin_amdgcn_global_load_async_to_lds_b128(
          (gv4i_p)(void*)apg[it], (lv4i_p)(void*)&As[buf][aofs[it]], 0, 0);
#else
      uint4 d = *(const uint4*)apg[it];
      unsigned short* dst = &As[buf][aofs[it]];
      *(uint2*)(dst + 0) = make_uint2(d.x, d.y);
      *(uint2*)(dst + 4) = make_uint2(d.z, d.w);
#endif
      apg[it] += BK;
    }
    #pragma unroll
    for (int it = 0; it < BITS; ++it) {
      float4 f = *(const float4*)bpg[it];
      int c = bcol[it], r = brow[it];
      Bs[buf][(c + 0) * BSTR + r] = f2bfu(f.x);
      Bs[buf][(c + 1) * BSTR + r] = f2bfu(f.y);
      Bs[buf][(c + 2) * BSTR + r] = f2bfu(f.z);
      Bs[buf][(c + 3) * BSTR + r] = f2bfu(f.w);
      __builtin_prefetch(bpg[it] + (size_t)BK * N, 0, 3);  // speculative: safe OOB
      bpg[it] += (size_t)BK * N;
    }
  };

  v8f acc[MI][4] = {};
  const int lrow = lane & 15;
  const int khalf = (lane >> 4) << 3;   // 0 or 8

  stage(0);
  const int nsteps = K >> 5;
  int cur = 0;
  for (int s = 0; s < nsteps; ++s, cur ^= 1) {
    wait_async_then_barrier();          // my async writes done, then block barrier
    if (s + 1 < nsteps) stage(cur ^ 1); // overlap next-tile loads with WMMA below

    v16bf af[MI], bfr[4];
    #pragma unroll
    for (int mi = 0; mi < MI; ++mi) {
      const unsigned short* ap = &As[cur][(wm * MI * 16 + mi * 16 + lrow) * ASTR + khalf];
      #pragma unroll
      for (int i = 0; i < 8; ++i) {
        af[mi][i]     = bfbits(ap[i]);
        af[mi][8 + i] = bfbits(ap[16 + i]);
      }
    }
    #pragma unroll
    for (int ni = 0; ni < 4; ++ni) {
      const unsigned short* bp = &Bs[cur][(wn * 64 + ni * 16 + lrow) * BSTR + khalf];
      #pragma unroll
      for (int i = 0; i < 8; ++i) {
        bfr[ni][i]     = bfbits(bp[i]);
        bfr[ni][8 + i] = bfbits(bp[16 + i]);
      }
    }
    #pragma unroll
    for (int mi = 0; mi < MI; ++mi)
      #pragma unroll
      for (int ni = 0; ni < 4; ++ni)
        acc[mi][ni] = __builtin_amdgcn_wmma_f32_16x16x32_bf16(
            false, af[mi], false, bfr[ni], (short)0, acc[mi][ni], false, false);
  }

  // epilogue (C/D layout: lane<16 -> M=r, N=lane; lane>=16 -> M=r+8, N=lane-16)
  const int ncol = lane & 15;
  const int moff = (lane >> 4) * 8;
  #pragma unroll
  for (int mi = 0; mi < MI; ++mi) {
    #pragma unroll
    for (int ni = 0; ni < 4; ++ni) {
      int gc = col0 + wn * 64 + ni * 16 + ncol;
      float bv = (flags & FLAG_BIAS) ? bias[gc] : 0.0f;
      #pragma unroll
      for (int r = 0; r < 8; ++r) {
        int gr = row0 + wm * MI * 16 + mi * 16 + moff + r;
        float val = acc[mi][ni][r] + bv;
        if (flags & FLAG_GELU) val = gelu_tanh(val);
        size_t off = (size_t)gr * N + gc;
        if (flags & FLAG_OBF) {
          Ch[(size_t)blockIdx.z * sC + off] = f2bfu(val);
        } else {
          if (flags & FLAG_RES) val += Res[off];
          Cf[(size_t)blockIdx.z * sC + off] = val;
        }
      }
    }
  }
}

// ---------------------------------------------------------------------------
static inline void launch_gemm(int MI, const unsigned short* A, unsigned long long sA,
                               const float* Bm, unsigned long long sB,
                               const float* bias, const float* res,
                               float* Cf, unsigned short* Ch, unsigned long long sC,
                               int M, int N, int K, int batch, int flags,
                               hipStream_t stream) {
  dim3 grid(N / BN, M / (64 * MI), batch);
  if (MI == 4)
    gemm_bf16_wmma<4><<<grid, 256, 0, stream>>>(A, sA, Bm, sB, bias, res, Cf, Ch, sC, N, K, flags);
  else
    gemm_bf16_wmma<2><<<grid, 256, 0, stream>>>(A, sA, Bm, sB, bias, res, Cf, Ch, sC, N, K, flags);
}

extern "C" void kernel_launch(void* const* d_in, const int* in_sizes, int n_in,
                              void* d_out, int out_size, void* d_ws, size_t ws_size,
                              hipStream_t stream) {
  (void)in_sizes; (void)n_in; (void)out_size; (void)ws_size;
  const int*   ids  = (const int*)d_in[0];
  const float* emb  = (const float*)d_in[1];
  const float* Wv   = (const float*)d_in[2];
  const float* Wo   = (const float*)d_in[3];
  const float* P    = (const float*)d_in[4];
  const float* Q    = (const float*)d_in[5];
  const float* ln1w = (const float*)d_in[6];
  const float* ln1b = (const float*)d_in[7];
  const float* ln2w = (const float*)d_in[8];
  const float* ln2b = (const float*)d_in[9];
  const float* w1   = (const float*)d_in[10];
  const float* b1   = (const float*)d_in[11];
  const float* w2   = (const float*)d_in[12];
  const float* b2   = (const float*)d_in[13];
  const float* lnfw = (const float*)d_in[14];
  const float* lnfb = (const float*)d_in[15];
  const float* hw   = (const float*)d_in[16];
  const float* hb   = (const float*)d_in[17];
  float* out = (float*)d_out;

  const int Bsz = 4, S = 512, D = 1024, L = 8, R = 64, H = 4096, V = 32000;
  const int SMAX = 512;
  const int Mr = Bsz * S;                 // 2048 token rows
  const size_t TD = (size_t)Mr * D;       // 2M elements

  // Workspace carve-up (~41 MB total)
  float* x = (float*)d_ws;                               // (Mr,D) f32 residual
  float* v = x + TD;                                     // (Mr,D) f32
  unsigned short* h_bf    = (unsigned short*)(v + TD);   // (Mr,D) bf16 LN out
  unsigned short* attn_bf = h_bf + TD;                   // (Mr,D) bf16
  unsigned short* t_bf    = attn_bf + TD;                // (Mr,H) bf16
  unsigned short* Mm_bf   = t_bf + (size_t)Mr * H;       // (S,S)  bf16

  embed_kernel<<<Mr, 256, 0, stream>>>(ids, emb, x, D);

  for (int l = 0; l < L; ++l) {
    const float* Wvl = Wv + (size_t)l * D * D;
    const float* Wol = Wo + (size_t)l * D * D;

    // h = LN1(x) ; v = h @ Wv (f32 out: consumed as GEMM-B next)
    layernorm_bf16_kernel<<<Mr, 256, 0, stream>>>(x, ln1w + l * D, ln1b + l * D, h_bf);
    launch_gemm(2, h_bf, 0, Wvl, 0, nullptr, nullptr, v, nullptr, 0,
                Mr, D, D, 1, 0, stream);

    // M = (P Q^T) * tril ; attn_b = M @ v_b  (batched, shared bf16 A)
    pqmask_kernel<<<dim3(S / 16, S / 16), 256, 0, stream>>>(
        P + (size_t)l * SMAX * R, Q + (size_t)l * SMAX * R, Mm_bf, S, R);
    launch_gemm(2, Mm_bf, 0, v, (unsigned long long)S * D, nullptr, nullptr,
                nullptr, attn_bf, (unsigned long long)S * D,
                S, D, S, Bsz, FLAG_OBF, stream);

    // x = x + attn @ Wo   (residual fused; in-place safe: 1 thread/elem)
    launch_gemm(2, attn_bf, 0, Wol, 0, nullptr, x, x, nullptr, 0,
                Mr, D, D, 1, FLAG_RES, stream);

    // MLP: t = gelu(LN2(x) @ w1 + b1) [bf16] ; x = x + t @ w2 + b2
    layernorm_bf16_kernel<<<Mr, 256, 0, stream>>>(x, ln2w + l * D, ln2b + l * D, h_bf);
    launch_gemm(4, h_bf, 0, w1 + (size_t)l * D * H, 0, b1 + (size_t)l * H, nullptr,
                nullptr, t_bf, 0, Mr, H, D, 1, FLAG_BIAS | FLAG_GELU | FLAG_OBF, stream);
    launch_gemm(2, t_bf, 0, w2 + (size_t)l * H * D, 0, b2 + (size_t)l * D, x,
                x, nullptr, 0, Mr, D, H, 1, FLAG_BIAS | FLAG_RES, stream);
  }

  // logits = LN_f(x) @ head_w + head_b
  layernorm_bf16_kernel<<<Mr, 256, 0, stream>>>(x, lnfw, lnfb, h_bf);
  launch_gemm(4, h_bf, 0, hw, 0, hb, nullptr, out, nullptr, 0,
              Mr, V, D, 1, FLAG_BIAS, stream);
}
